// GIN_Cross_Attention_Net_41618233099061
// MI455X (gfx1250) — matrix operand order repaired and checked by reading, not compiled, not run on hardware
//
#include <hip/hip_runtime.h>

// ---------------- problem constants (match reference) ----------------
#define NTOT  32768      // B*L nodes
#define ETOT  131072     // edges
#define BGR   512        // graphs
#define LGR   64         // nodes per graph
#define HIDC  512
#define ATTNC 128
#define NODEC 128
#define NLAYERS 5

typedef __attribute__((ext_vector_type(16))) __bf16 v16bf;
typedef __attribute__((ext_vector_type(8)))  __bf16 v8bf;
typedef __attribute__((ext_vector_type(8)))  float  v8f;

// ---------------- elementwise helpers ----------------
__global__ __launch_bounds__(256) void k_cvt_bf16(const float* __restrict__ s,
                                                  __bf16* __restrict__ d, int n) {
  int i = blockIdx.x * 256 + threadIdx.x;
  if (i < n) d[i] = (__bf16)s[i];
}

__global__ __launch_bounds__(256) void k_add_cvt_bf16(const float* __restrict__ a,
                                                      const float* __restrict__ b,
                                                      __bf16* __restrict__ d, int n) {
  int i = blockIdx.x * 256 + threadIdx.x;
  if (i < n) d[i] = (__bf16)(a[i] + b[i]);
}

__global__ __launch_bounds__(256) void k_zero(float* __restrict__ p, int n) {
  int i = blockIdx.x * 256 + threadIdx.x;
  if (i < n) p[i] = 0.f;
}

// ---------------- weight swizzle: f32 [K,N] -> fragment-ready bf16 ----------------
// Wsw element index i = ((ks*NT + nt)*32 + lane)*16 + e  maps to
// W[32*ks + 16*(lane>>4) + e][16*nt + (lane&15)]  (contiguous k order, same as the A fragment:
// lanes 0-15 hold K=0..15, lanes 16-31 hold K=16..31).
// After this, a lane's whole 16-element B fragment is one contiguous 32B run.
__global__ __launch_bounds__(256)
void k_swizzle_w(const float* __restrict__ W, __bf16* __restrict__ Wsw, int K, int N) {
  int i = blockIdx.x * 256 + threadIdx.x;
  if (i >= K * N) return;
  const int e    = i & 15;
  const int lane = (i >> 4) & 31;
  const int t    = i >> 9;              // ks*NT + nt
  const int NT   = N >> 4;
  const int nt   = t % NT;
  const int ks   = t / NT;
  const int k = 32 * ks + 16 * (lane >> 4) + e;
  const int n = 16 * nt + (lane & 15);
  Wsw[i] = (__bf16)W[(size_t)k * N + n];
}

// ---------------- WMMA bf16 GEMM: C[M,Ncol] = act(A[M,KC]@W + bias) ----------------
// Block: 256 threads = 8 waves; block tile 128(M) x 128(N). Wave w owns rows
// 16w..16w+15 and 8 16x16 n-tiles (8 wmma per A fragment). K templated so the
// step loop fully unrolls. A tile double-buffered in LDS: tile ks+1 prefetched
// into registers during compute on tile ks, one barrier per step. B fragments
// come straight from the pre-swizzled weight (32B contiguous per lane,
// coalesced, L2/WGP$-resident).
template <int KC>
__global__ __launch_bounds__(256)
void k_gemm_bf16(const __bf16* __restrict__ A, const v16bf* __restrict__ Wsw,
                 const float* __restrict__ bias, float* __restrict__ C,
                 int Ncol, int relu)
{
  constexpr int NSTEP = KC / 32;
  __shared__ __align__(16) __bf16 sA[2][128][40];  // 80B row stride: aligned + bank-skewed

  const int tid  = threadIdx.x;
  const int wave = tid >> 5;
  const int lane = tid & 31;
  const int row0 = blockIdx.x * 128;
  const int col0 = blockIdx.y * 128;
  const int mrow = lane & 15;
  const int half = lane >> 4;
  const int NT   = Ncol >> 4;            // 16-wide column tiles in W
  const int nt0  = col0 >> 4;            // first n-tile of this block

  const v8f vzero = {0.f,0.f,0.f,0.f,0.f,0.f,0.f,0.f};
  v8f acc[8];
#pragma unroll
  for (int i = 0; i < 8; ++i) acc[i] = vzero;

  const int ar = tid >> 1, ah = tid & 1;  // A loaders: 2 x uint4 per thread per step
  const uint4* agp = reinterpret_cast<const uint4*>(A + (size_t)(row0 + ar) * KC + 16 * ah);

  // prologue: stage tile 0
  {
    uint4* dp = reinterpret_cast<uint4*>(&sA[0][ar][16 * ah]);
    dp[0] = agp[0]; dp[1] = agp[1];
  }
  __syncthreads();

#pragma unroll
  for (int ks = 0; ks < NSTEP; ++ks) {
    const int buf = ks & 1;

    // prefetch next A tile into registers (no LDS/barrier dependency)
    uint4 pf0, pf1;
    if (ks + 1 < NSTEP) {
      const uint4* sp = agp + (size_t)(ks + 1) * 2;  // +16 bf16 = 2 uint4 along K
      pf0 = sp[0]; pf1 = sp[1];
    }

    // A fragment (16x32 bf16): element e -> K=16*half+e (contiguous, matches Wsw)
    const __bf16* arow = &sA[buf][16 * wave + mrow][0];
    const v8bf lo = *reinterpret_cast<const v8bf*>(arow + 16 * half);
    const v8bf hi = *reinterpret_cast<const v8bf*>(arow + 16 * half + 8);
    const v16bf af = __builtin_shufflevector(lo, hi, 0,1,2,3,4,5,6,7,8,9,10,11,12,13,14,15);

#pragma unroll
    for (int nt = 0; nt < 8; ++nt) {
      const v16bf bfr = Wsw[(size_t)(ks * NT + nt0 + nt) * 32 + lane];  // 32B contiguous
      acc[nt] = __builtin_amdgcn_wmma_f32_16x16x32_bf16(
          /*neg_a=*/false, af, /*neg_b=*/false, bfr,
          /*c_mod=*/(short)0, acc[nt], /*reuse_a=*/false, /*reuse_b=*/false);
    }

    // stage next tile into the alternate buffer; single barrier per step
    if (ks + 1 < NSTEP) {
      uint4* dp = reinterpret_cast<uint4*>(&sA[buf ^ 1][ar][16 * ah]);
      dp[0] = pf0; dp[1] = pf1;
      __syncthreads();
    }
  }

  // epilogue: D layout — lane holds N = lane&15, M = r + 8*half in vgpr r
#pragma unroll
  for (int nt = 0; nt < 8; ++nt) {
    const int n = col0 + nt * 16 + mrow;
    const float bv = bias[n];
#pragma unroll
    for (int r = 0; r < 8; ++r) {
      const int mm = row0 + 16 * wave + r + 8 * half;
      float o = acc[nt][r] + bv;
      if (relu) o = fmaxf(o, 0.f);
      C[(size_t)mm * Ncol + n] = o;
    }
  }
}

// ---------------- fused GINEConv message pass ----------------
// one wave per edge: e = ef[e]@We + be (We cached in LDS), m = relu(hid[src]+e),
// scatter-add into aggr[dst] via global_atomic_add_f32.
__global__ __launch_bounds__(256)
void k_edge_scatter(const float* __restrict__ ef, const float* __restrict__ We,
                    const float* __restrict__ be, const float* __restrict__ hid,
                    const int* __restrict__ src, const int* __restrict__ dst,
                    float* __restrict__ aggr, int E_)
{
  __shared__ float sW[16 * HIDC];   // 32 KB layer weight
  __shared__ float sb[HIDC];        //  2 KB layer bias
  for (int i = threadIdx.x; i < 16 * HIDC; i += 256) sW[i] = We[i];
  for (int i = threadIdx.x; i < HIDC; i += 256) sb[i] = be[i];
  __syncthreads();

  const int wave = threadIdx.x >> 5, lane = threadIdx.x & 31;
  const int e = blockIdx.x * 8 + wave;
  if (e >= E_) return;
  const int s = src[e], d = dst[e];

  float f[16];
#pragma unroll
  for (int k = 0; k < 16; ++k) f[k] = ef[(size_t)e * 16 + k];

  for (int c = lane; c < HIDC; c += 32) {     // coalesced across the wave
    float a = sb[c];
#pragma unroll
    for (int k = 0; k < 16; ++k) a = fmaf(f[k], sW[k * HIDC + c], a);
    float m = hid[(size_t)s * HIDC + c] + a;
    m = fmaxf(m, 0.f);
    atomicAdd(&aggr[(size_t)d * HIDC + c], m);
  }
}

// ---------------- per-graph attention + global_add_pool ----------------
__global__ __launch_bounds__(64)
void k_attn_pool(const float* __restrict__ q, const float* __restrict__ k,
                 const float* __restrict__ v, float* __restrict__ out)
{
  __shared__ float sK[LGR * ATTNC];   // 32 KB (scores source, then ca staging)
  __shared__ float sV[LGR * ATTNC];   // 32 KB
  const int b = blockIdx.x, t = threadIdx.x;
  const size_t base = (size_t)b * LGR * ATTNC;

  for (int i = t; i < LGR * ATTNC; i += 64) { sK[i] = k[base + i]; sV[i] = v[base + i]; }
  __syncthreads();

  float sc[LGR];
#pragma unroll
  for (int j = 0; j < LGR; ++j) sc[j] = 0.f;
  for (int d0 = 0; d0 < ATTNC; d0 += 16) {    // tile q row to bound VGPRs
    float qv[16];
#pragma unroll
    for (int j = 0; j < 16; ++j) qv[j] = q[base + (size_t)t * ATTNC + d0 + j];
#pragma unroll
    for (int kk = 0; kk < LGR; ++kk) {
      float a = sc[kk];
#pragma unroll
      for (int j = 0; j < 16; ++j) a = fmaf(qv[j], sK[kk * ATTNC + d0 + j], a);
      sc[kk] = a;
    }
  }
  const float scale = 0.08838834764831845f;   // 1/sqrt(128)
  float mx = -3.4e38f;
#pragma unroll
  for (int j = 0; j < LGR; ++j) { sc[j] *= scale; mx = fmaxf(mx, sc[j]); }
  float sum = 0.f;
#pragma unroll
  for (int j = 0; j < LGR; ++j) { sc[j] = __expf(sc[j] - mx); sum += sc[j]; }
  const float inv = 1.f / sum;
  __syncthreads();                             // sK free after scores

  for (int d = 0; d < ATTNC; ++d) {
    float a = 0.f;
#pragma unroll
    for (int kk = 0; kk < LGR; ++kk) a = fmaf(sc[kk], sV[kk * ATTNC + d], a);
    sK[t * ATTNC + d] = a * inv;
  }
  __syncthreads();

  for (int d = t; d < ATTNC; d += 64) {        // fixed-order column sums -> deterministic
    float a = 0.f;
    for (int r = 0; r < LGR; ++r) a += sK[r * ATTNC + d];
    out[(size_t)b * ATTNC + d] = a;
  }
}

// ---------------- orchestration ----------------
extern "C" void kernel_launch(void* const* d_in, const int* in_sizes, int n_in,
                              void* d_out, int out_size, void* d_ws, size_t ws_size,
                              hipStream_t stream)
{
  (void)in_sizes; (void)n_in; (void)out_size; (void)ws_size;
  const float* node_feature = (const float*)d_in[0];
  const float* edge_feature = (const float*)d_in[1];
  const float* w_n2h = (const float*)d_in[2];
  const float* b_n2h = (const float*)d_in[3];
  const float* w_edge = (const float*)d_in[4];   // [5,16,512]
  const float* b_edge = (const float*)d_in[5];   // [5,512]
  const float* w_nn = (const float*)d_in[6];
  const float* b_nn = (const float*)d_in[7];
  const float* w1 = (const float*)d_in[8];
  const float* b1 = (const float*)d_in[9];
  const float* w2 = (const float*)d_in[10];
  const float* b2 = (const float*)d_in[11];
  const float* wq = (const float*)d_in[12];
  const float* bq = (const float*)d_in[13];
  const float* wk = (const float*)d_in[14];
  const float* bk = (const float*)d_in[15];
  const float* wv = (const float*)d_in[16];
  const float* bv = (const float*)d_in[17];
  const int* edge_index = (const int*)d_in[18];
  const int* esrc = edge_index;
  const int* edst = edge_index + ETOT;
  float* out = (float*)d_out;

  // workspace carve-out (~163 MB)
  char* wsb = (char*)d_ws; size_t off = 0;
  auto take = [&](size_t bytes) -> char* {
    char* p = wsb + off; off += (bytes + 255) & ~(size_t)255; return p;
  };
  __bf16* Abf = (__bf16*)take((size_t)NTOT * HIDC * 2);   // bf16 A-operand staging
  float*  H   = (float*) take((size_t)NTOT * HIDC * 4);   // hid / out
  float*  G   = (float*) take((size_t)NTOT * HIDC * 4);   // aggr / t1 / qkv
  __bf16* wb_n2h = (__bf16*)take((size_t)NODEC * HIDC * 2);
  __bf16* wb_nn  = (__bf16*)take((size_t)HIDC * HIDC * 2);
  __bf16* wb_1   = (__bf16*)take((size_t)HIDC * HIDC * 2);
  __bf16* wb_2   = (__bf16*)take((size_t)HIDC * HIDC * 2);
  __bf16* wb_q   = (__bf16*)take((size_t)HIDC * ATTNC * 2);
  __bf16* wb_k   = (__bf16*)take((size_t)HIDC * ATTNC * 2);
  __bf16* wb_v   = (__bf16*)take((size_t)HIDC * ATTNC * 2);

  auto cvt = [&](const float* s, __bf16* d, int n) {
    k_cvt_bf16<<<(n + 255) / 256, 256, 0, stream>>>(s, d, n);
  };
  auto swz = [&](const float* W, __bf16* Wsw, int K, int N) {
    k_swizzle_w<<<(K * N + 255) / 256, 256, 0, stream>>>(W, Wsw, K, N);
  };
  auto gemm = [&](const __bf16* Am, const __bf16* Wm, const float* bias,
                  float* Cm, int K, int Ncol, int relu) {
    dim3 g(NTOT / 128, Ncol / 128);
    if (K == 128)
      k_gemm_bf16<128><<<g, 256, 0, stream>>>(Am, (const v16bf*)Wm, bias, Cm, Ncol, relu);
    else
      k_gemm_bf16<512><<<g, 256, 0, stream>>>(Am, (const v16bf*)Wm, bias, Cm, Ncol, relu);
  };

  // weights -> fragment-ready bf16 (one-time, ~2 MB total)
  swz(w_n2h, wb_n2h, NODEC, HIDC);
  swz(w_nn,  wb_nn,  HIDC,  HIDC);
  swz(w1,    wb_1,   HIDC,  HIDC);
  swz(w2,    wb_2,   HIDC,  HIDC);
  swz(wq,    wb_q,   HIDC,  ATTNC);
  swz(wk,    wb_k,   HIDC,  ATTNC);
  swz(wv,    wb_v,   HIDC,  ATTNC);

  // hid = node @ w_n2h + b   (no relu)
  cvt(node_feature, Abf, NTOT * NODEC);
  gemm(Abf, wb_n2h, b_n2h, H, NODEC, HIDC, 0);

  // 5 x GINEConv
  for (int l = 0; l < NLAYERS; ++l) {
    k_zero<<<(NTOT * HIDC + 255) / 256, 256, 0, stream>>>(G, NTOT * HIDC);
    k_edge_scatter<<<ETOT / 8, 256, 0, stream>>>(
        edge_feature, w_edge + (size_t)l * 16 * HIDC, b_edge + (size_t)l * HIDC,
        H, esrc, edst, G, ETOT);
    k_add_cvt_bf16<<<(NTOT * HIDC + 255) / 256, 256, 0, stream>>>(H, G, Abf, NTOT * HIDC);
    gemm(Abf, wb_nn, b_nn, H, HIDC, HIDC, 1);        // relu((hid+aggr)@w_nn+b)
  }

  // MLP head: t1 = relu(hid@w1+b1); out = t1@w2+b2
  cvt(H, Abf, NTOT * HIDC);
  gemm(Abf, wb_1, b1, G, HIDC, HIDC, 1);
  cvt(G, Abf, NTOT * HIDC);
  gemm(Abf, wb_2, b2, H, HIDC, HIDC, 0);

  // q/k/v projections (out -> bf16 once, reuse for all three)
  cvt(H, Abf, NTOT * HIDC);
  float* Q  = G;
  float* Kt = G + (size_t)NTOT * ATTNC;
  float* V  = G + (size_t)2 * NTOT * ATTNC;
  gemm(Abf, wb_q, bq, Q,  HIDC, ATTNC, 0);
  gemm(Abf, wb_k, bk, Kt, HIDC, ATTNC, 0);
  gemm(Abf, wb_v, bv, V,  HIDC, ATTNC, 0);

  // attention + global_add_pool -> [512,128]
  k_attn_pool<<<BGR, 64, 0, stream>>>(Q, Kt, V, out);
}
